// Att_mask_47571057771085
// MI455X (gfx1250) — matrix operand order repaired and checked by reading, not compiled, not run on hardware
//
#include <hip/hip_runtime.h>
#include <hip/hip_bf16.h>

typedef __attribute__((ext_vector_type(16))) _Float16 v16h;
typedef __attribute__((ext_vector_type(8)))  _Float16 v8h;
typedef __attribute__((ext_vector_type(8)))  float    v8f;

#define N_AGENTS 16
#define D_DIM    64
#define R_DIM    32
#define WAVES_PER_BLOCK 8
#define BLOCK_THREADS (WAVES_PER_BLOCK * 32)
#define NUM_BLOCKS 2048

__device__ __forceinline__ float halfwave_sum16(float v) {
    // reduce across the 16 lanes of each half-wave (masks < 16 never cross halves)
    v += __shfl_xor(v, 1, 32);
    v += __shfl_xor(v, 2, 32);
    v += __shfl_xor(v, 4, 32);
    v += __shfl_xor(v, 8, 32);
    return v;
}

// gfx1250 hardware transcendental V_TANH_F32 (probe-confirmed).
#if __has_builtin(__builtin_amdgcn_tanhf)
__device__ __forceinline__ float fast_tanh(float x) {
    return __builtin_amdgcn_tanhf(x);
}
#else
__device__ __forceinline__ float fast_tanh(float x) {
    float ax = __builtin_fabsf(x);
    float e  = __expf(-2.0f * ax);
    float r  = (1.0f - e) * __builtin_amdgcn_rcpf(1.0f + e);
    return __builtin_copysignf(r, x);
}
#endif

// Native base-2 exponential (v_exp_f32).
#if __has_builtin(__builtin_amdgcn_exp2f)
__device__ __forceinline__ float fast_exp2(float x) { return __builtin_amdgcn_exp2f(x); }
#else
__device__ __forceinline__ float fast_exp2(float x) { return exp2f(x); }
#endif

// One 16x16 output tile of (A[32,64] @ x[64,16]) via two chained K=32 WMMAs.
__device__ __forceinline__ v8f proj_tile(const _Float16* __restrict__ aw,
                                         const v16h* bx, int mi, int lm, int hw) {
    v8f acc = {};
#pragma unroll
    for (int ki = 0; ki < 2; ++ki) {
        // A-layout: lane row = mi*16+lm; elem j -> K = hw*16+j
        const _Float16* arow = aw + (mi * 16 + lm) * D_DIM + ki * 32;
        v8h lo = *(const v8h*)(arow + hw * 16);
        v8h hi = *(const v8h*)(arow + hw * 16 + 8);
        v16h a;
#pragma unroll
        for (int j = 0; j < 8; ++j) { a[j] = lo[j]; a[j + 8] = hi[j]; }
        acc = __builtin_amdgcn_wmma_f32_16x16x32_f16(
            false, a, false, bx[ki], (short)0, acc, false, false);
    }
    return acc;
}

__global__ __launch_bounds__(BLOCK_THREADS)
void att_mask_kernel(const float* __restrict__ x,
                     const float* __restrict__ L,
                     const float* __restrict__ Aq,
                     const float* __restrict__ Ak,
                     const float* __restrict__ Av,
                     const float* __restrict__ Ao,
                     float* __restrict__ out,
                     int batch)
{
    // Block-shared f16 copies of the three projection matrices: [3][32][64]
    __shared__ _Float16 sA[3 * R_DIM * D_DIM];                    // 12 KB
    // Per-wave scratch:
    __shared__ _Float16 sQ[WAVES_PER_BLOCK][R_DIM * N_AGENTS];    //  8 KB  Q[r][n]
    __shared__ _Float16 sKpad[WAVES_PER_BLOCK][R_DIM * R_DIM];    // 16 KB  K[s][k], k>=16 zero pad
    __shared__ _Float16 sVt[WAVES_PER_BLOCK][N_AGENTS * R_DIM];   //  8 KB  Vt[n][r] (transposed)
    __shared__ _Float16 sP[WAVES_PER_BLOCK][R_DIM * R_DIM];       // 16 KB  attn[r][s]

    const int tid = threadIdx.x;

    // ---- one-time: convert Aq/Ak/Av f32 -> f16 into LDS ----
    for (int idx = tid; idx < R_DIM * D_DIM; idx += BLOCK_THREADS) {
        sA[idx]                     = (_Float16)Aq[idx];
        sA[R_DIM * D_DIM + idx]     = (_Float16)Ak[idx];
        sA[2 * R_DIM * D_DIM + idx] = (_Float16)Av[idx];
    }
    __syncthreads();

    const int lane = tid & 31;
    const int hw   = lane >> 4;   // half-wave id (0/1)
    const int lm   = lane & 15;   // lane-in-half (row/col index)
    const int wib  = tid >> 5;    // wave in block
    const int gwave  = blockIdx.x * WAVES_PER_BLOCK + wib;
    const int nwaves = gridDim.x * WAVES_PER_BLOCK;

    _Float16* myQ = &sQ[wib][0];
    _Float16* myK = &sKpad[wib][0];
    _Float16* myV = &sVt[wib][0];
    _Float16* myP = &sP[wib][0];

    // ---- one-time: zero K pad region (per-iteration writes only touch k<16) ----
    {
        unsigned* kz = (unsigned*)myK;
#pragma unroll
        for (int t = 0; t < (R_DIM * R_DIM / 2) / 32; ++t)
            kz[t * 32 + lane] = 0u;
    }

    // Ao coefficients for this lane's accumulator rows: r = mi*16 + v + 8*hw
    float ao[2][8];
#pragma unroll
    for (int mi = 0; mi < 2; ++mi)
#pragma unroll
        for (int v = 0; v < 8; ++v)
            ao[mi][v] = Ao[mi * 16 + v + 8 * hw];

    for (int b = gwave; b < batch; b += nwaves) {
        // ---- numNeighbors from first row of L[b]: ballot + popcount (SALU) ----
        float lv = L[(size_t)b * (N_AGENTS * N_AGENTS) + lm];  // lanes 16-31 mirror 0-15
        unsigned long long m = __ballot(lv >= 1.0f);
        float nb = (float)__popc((unsigned)m & 0xFFFFu);
        const float scale = __builtin_amdgcn_rsqf(nb + 1.0f);
        const float scale2 = scale * 1.4426950408889634f;  // fold log2(e) for exp2-based softmax

        // ---- load x[b] as two 32x16 WMMA B-tiles (f32 -> f16) ----
        // B-layout: lane holds column n=lm; element j <-> K = hw*16 + j
        const float* xb = x + (size_t)b * (D_DIM * N_AGENTS);
        v16h bx[2];
#pragma unroll
        for (int ki = 0; ki < 2; ++ki)
#pragma unroll
            for (int j = 0; j < 16; ++j)
                bx[ki][j] = (_Float16)xb[(ki * 32 + hw * 16 + j) * N_AGENTS + lm];

        // ---- Q/K/V = tanh(A @ x), 12 WMMAs ----
        // D-layout: lane holds D[mi*16 + v + 8*hw][lm] in acc element v.
#pragma unroll
        for (int mi = 0; mi < 2; ++mi) {
            // Q -> row-major [r][n]
            v8f q = proj_tile(sA, bx, mi, lm, hw);
#pragma unroll
            for (int v = 0; v < 8; ++v)
                myQ[(mi * 16 + v + 8 * hw) * N_AGENTS + lm] = (_Float16)fast_tanh(q[v]);

            // K -> zero-padded [s][k] with row stride 32 (pad k>=16 stays zero)
            v8f k = proj_tile(sA + R_DIM * D_DIM, bx, mi, lm, hw);
#pragma unroll
            for (int v = 0; v < 8; ++v)
                myK[(mi * 16 + v + 8 * hw) * R_DIM + lm] = (_Float16)fast_tanh(k[v]);

            // V -> transposed [n][r]: lane's 8 rows are contiguous -> one b128 store
            v8f vv = proj_tile(sA + 2 * R_DIM * D_DIM, bx, mi, lm, hw);
            v8h vp;
#pragma unroll
            for (int v = 0; v < 8; ++v)
                vp[v] = (_Float16)fast_tanh(vv[v]);
            *(v8h*)(&myV[lm * R_DIM + mi * 16 + 8 * hw]) = vp;
        }

        // ---- scores = Q @ K^T (K-dim = n = 16, zero-padded to 32) : 4 WMMAs ----
        v8f sc[2][2];
#pragma unroll
        for (int ri = 0; ri < 2; ++ri) {
#pragma unroll
            for (int si = 0; si < 2; ++si) {
                const _Float16* qrow = &myQ[(ri * 16 + lm) * N_AGENTS];
                v8h ql = *(const v8h*)(qrow + hw * 8);
                v16h a;
#pragma unroll
                for (int j = 0; j < 8; ++j) { a[j] = ql[j]; a[j + 8] = (_Float16)0.0f; }
                // B = K^T from zero-padded tile: element j<8 <-> k = hw*8 + j, j>=8 <-> k = 16 + hw*8 + (j-8), column s = lm
                const _Float16* krow = &myK[(si * 16 + lm) * R_DIM + hw * 8];
                v8h k0 = *(const v8h*)(krow);
                v8h k1 = *(const v8h*)(krow + 16);
                v16h bb;
#pragma unroll
                for (int j = 0; j < 8; ++j) { bb[j] = k0[j]; bb[j + 8] = k1[j]; }
                v8f c = {};
                sc[ri][si] = __builtin_amdgcn_wmma_f32_16x16x32_f16(
                    false, a, false, bb, (short)0, c, false, false);
            }
        }

        // ---- softmax over s (row spread across si tiles + 16 lanes of a half-wave) ----
        // |scores| <= 16 (tanh-bounded Q,K), so exp cannot overflow: no max-subtract needed.
        // exp(x*scale) computed as exp2(x*(scale*log2e)) -> single mul + v_exp_f32.
#pragma unroll
        for (int ri = 0; ri < 2; ++ri) {
#pragma unroll
            for (int v = 0; v < 8; ++v) {
                float e0 = fast_exp2(sc[ri][0][v] * scale2);
                float e1 = fast_exp2(sc[ri][1][v] * scale2);
                float sm = halfwave_sum16(e0 + e1);
                float inv = __builtin_amdgcn_rcpf(sm);
                const int r = ri * 16 + v + 8 * hw;
                myP[r * R_DIM + lm]      = (_Float16)(e0 * inv);
                myP[r * R_DIM + 16 + lm] = (_Float16)(e1 * inv);
            }
        }

        // ---- o = attn @ V (K=32) : 2 WMMAs, fused with R = tanh(Ao @ o) ----
        v16h vb;  // B = V: column n=lm, element j <-> K = hw*16 + j; contiguous in Vt[n][r]
        {
            const _Float16* vrow = &myV[lm * R_DIM + hw * 16];
            v8h v0 = *(const v8h*)(vrow);
            v8h v1 = *(const v8h*)(vrow + 8);
#pragma unroll
            for (int j = 0; j < 8; ++j) { vb[j] = v0[j]; vb[j + 8] = v1[j]; }
        }

        float part = 0.0f;
#pragma unroll
        for (int mi = 0; mi < 2; ++mi) {
            const _Float16* prow = &myP[(mi * 16 + lm) * R_DIM];
            v8h p0 = *(const v8h*)(prow + hw * 16);
            v8h p1 = *(const v8h*)(prow + hw * 16 + 8);
            v16h a;
#pragma unroll
            for (int j = 0; j < 8; ++j) { a[j] = p0[j]; a[j + 8] = p1[j]; }
            v8f c = {};
            v8f o = __builtin_amdgcn_wmma_f32_16x16x32_f16(
                false, a, false, vb, (short)0, c, false, false);
#pragma unroll
            for (int v = 0; v < 8; ++v)
                part += ao[mi][v] * o[v];
        }
        // combine the two half-waves (rows differ by 8) -> full 32-term dot per n=lm
        float full = part + __shfl_xor(part, 16, 32);
        float R  = fast_tanh(full);
        float R2 = R * R;
        // Laplacian-like output: out[i,j] = (j==i) ? sum_j R2 : -R2[j]
        float tot = halfwave_sum16(R2);
        if (lane < 16) {
            const int i = b & 15;
            out[(size_t)b * N_AGENTS + lm] = (lm == i) ? tot : -R2;
        }
    }
}

extern "C" void kernel_launch(void* const* d_in, const int* in_sizes, int n_in,
                              void* d_out, int out_size, void* d_ws, size_t ws_size,
                              hipStream_t stream) {
    const float* x  = (const float*)d_in[0];
    const float* L  = (const float*)d_in[1];
    const float* Aq = (const float*)d_in[2];
    const float* Ak = (const float*)d_in[3];
    const float* Av = (const float*)d_in[4];
    const float* Ao = (const float*)d_in[5];
    float* out = (float*)d_out;

    const int batch = in_sizes[0] / (D_DIM * N_AGENTS);  // 65536

    att_mask_kernel<<<NUM_BLOCKS, BLOCK_THREADS, 0, stream>>>(
        x, L, Aq, Ak, Av, Ao, out, batch);
}
